// MambaEncoder_4896262718011
// MI455X (gfx1250) — compile-verified, matrix operand order checked by reading, not run on hardware
//
#include <hip/hip_runtime.h>
#include <hip/hip_bf16.h>
#include <math.h>

// Problem constants (match reference)
#define B_   4
#define S_   2048
#define D_   256
#define N_   16
#define DT_  16
#define K_   4
#define NL_  4
#define EPS_ 1e-5f
#define M_   (B_ * S_)   // 8192 rows
#define NC_  8           // scan chunks (S_/NC_ = 256 steps per chunk)
#define TCH  64          // LDS staging tile (timesteps)

typedef __attribute__((ext_vector_type(16))) __bf16 v16bf;
typedef __attribute__((ext_vector_type(8)))  float  v8f;

union FragU { uint4 q[2]; unsigned short us[16]; v16bf v; };

__device__ __forceinline__ unsigned short f2bf(float f) {
  unsigned int u = __float_as_uint(f);
  u += 0x7FFFu + ((u >> 16) & 1u);   // round-to-nearest-even
  return (unsigned short)(u >> 16);
}

// ---------------------------------------------------------------------------
// gfx1250 async global->LDS copy (ASYNCcnt path), guarded so the build never
// breaks if the toolchain lacks the builtins. The builtin takes b32-typed
// (int*) pointers in AS1 (global) / AS3 (LDS). LDS generic pointers: low 32
// bits of the flat address are the LDS byte offset (ISA aperture rule), so
// inttoptr into AS3 is valid; global generic VA inttoptrs into AS1 directly.
// ---------------------------------------------------------------------------
#if __has_builtin(__builtin_amdgcn_global_load_async_to_lds_b32) && \
    __has_builtin(__builtin_amdgcn_s_wait_asynccnt)
#define USE_ASYNC_LDS 1
#else
#define USE_ASYNC_LDS 0
#endif

typedef __attribute__((address_space(1))) int gas_i32;
typedef __attribute__((address_space(3))) int las_i32;

__device__ __forceinline__ gas_i32* as_global(const void* p) {
  return (gas_i32*)(unsigned long long)(uintptr_t)p;
}
__device__ __forceinline__ las_i32* as_lds(const void* p) {
  return (las_i32*)(uintptr_t)(unsigned int)(uintptr_t)p;
}

__device__ __forceinline__ void lds_cp32(float* dst, const float* src) {
#if USE_ASYNC_LDS
  __builtin_amdgcn_global_load_async_to_lds_b32(as_global(src), as_lds(dst), 0, 0);
#else
  *dst = *src;
#endif
}
__device__ __forceinline__ void lds_cp_wait() {
#if USE_ASYNC_LDS
  __builtin_amdgcn_s_wait_asynccnt(0);
#endif
}

// ---------------------------------------------------------------------------
// bf16 fragment loader for CDNA5 WMMA 16x16x32 (compile-time K, no branches).
// A-layout (ISA 7.12.2): lanes 0-15 = rows, hold K = {k0..k0+7, k0+16..k0+23};
// lanes 16-31 = same rows, hold K = {k0+8..k0+15, k0+24..k0+31}.
// B fragment mirrors this with lanes = columns, so loading the rows of a
// row-major W[N][K] with the SAME loader yields B = W^T (we need X @ W^T).
// ---------------------------------------------------------------------------
template<int KV>
__device__ __forceinline__ v16bf load_frag(const unsigned short* __restrict__ base,
                                           int lane, int ld, int k0) {
  int r    = lane & 15;
  int koff = (lane & 16) ? 8 : 0;
  const unsigned short* p = base + (size_t)r * ld + k0 + koff;
  FragU f;
  f.q[0] = *(const uint4*)p;                        // global_load_b128
  if constexpr (KV % 32 == 0) {
    f.q[1] = *(const uint4*)(p + 16);               // global_load_b128
  } else {
    f.q[1].x = 0u; f.q[1].y = 0u; f.q[1].z = 0u; f.q[1].w = 0u;  // K 16..31 = 0
  }
  return f.v;
}

// ---------------------------------------------------------------------------
// WMMA GEMM: C[M x N] = A[M x K](bf16) @ W[N x K]^T(bf16), f32 accum.
// Each wave computes FOUR 16x16 M-tiles sharing one B fragment (64x16);
// 4 waves per block along N; compile-time K -> fully unrolled pipeline.
// MODE 0: plain (+optional bias, optional bf16 copy)
// MODE 1: split columns into u_pre / r (in_proj)
// MODE 2: softplus(acc + bias)  (dt_proj -> delta)
// ---------------------------------------------------------------------------
template<int MODE, int KV>
__global__ void gemm_wmma(const unsigned short* __restrict__ A, int lda,
                          const unsigned short* __restrict__ W,
                          int N,
                          const float* __restrict__ bias,
                          float* __restrict__ out0,
                          float* __restrict__ out1,
                          unsigned short* __restrict__ outb) {
  int wave = threadIdx.x >> 5;
  int lane = threadIdx.x & 31;
  int n0   = (blockIdx.x * (blockDim.x >> 5) + wave) * 16;
  int m0   = blockIdx.y * 64;
  if (n0 >= N) return;                         // whole-wave uniform exit

  v8f c0 = {}, c1 = {}, c2 = {}, c3 = {};
  const unsigned short* A0 = A + (size_t)m0 * lda;
  const unsigned short* Wn = W + (size_t)n0 * KV;
#pragma unroll
  for (int k0 = 0; k0 < KV; k0 += 32) {
    v16bf b  = load_frag<KV>(Wn, lane, KV, k0);
    v16bf a0 = load_frag<KV>(A0,                    lane, lda, k0);
    v16bf a1 = load_frag<KV>(A0 + (size_t)16 * lda, lane, lda, k0);
    v16bf a2 = load_frag<KV>(A0 + (size_t)32 * lda, lane, lda, k0);
    v16bf a3 = load_frag<KV>(A0 + (size_t)48 * lda, lane, lda, k0);
    c0 = __builtin_amdgcn_wmma_f32_16x16x32_bf16(false, a0, false, b, (short)0, c0, false, false);
    c1 = __builtin_amdgcn_wmma_f32_16x16x32_bf16(false, a1, false, b, (short)0, c1, false, false);
    c2 = __builtin_amdgcn_wmma_f32_16x16x32_bf16(false, a2, false, b, (short)0, c2, false, false);
    c3 = __builtin_amdgcn_wmma_f32_16x16x32_bf16(false, a3, false, b, (short)0, c3, false, false);
  }

  // C/D layout: VGPR j -> row +j (lanes 0-15) / +j+8 (lanes 16-31), col = lane&15
  int col = lane & 15;
  int rb  = (lane >> 4) * 8;
  int n   = n0 + col;
  float bv = (MODE == 2) ? bias[n] : ((MODE == 0 && bias) ? bias[n] : 0.f);
  v8f accs[4] = { c0, c1, c2, c3 };
#pragma unroll
  for (int mi = 0; mi < 4; mi++) {
#pragma unroll
    for (int j = 0; j < 8; j++) {
      int m   = m0 + mi * 16 + rb + j;
      float v = accs[mi][j];
      if (MODE == 0) {
        v += bv;
        out0[(size_t)m * N + n] = v;
        if (outb) outb[(size_t)m * N + n] = f2bf(v);
      } else if (MODE == 1) {
        if (n < D_) out0[(size_t)m * D_ + n] = v;
        else        out1[(size_t)m * D_ + (n - D_)] = v;
      } else { // MODE 2: softplus
        v += bv;
        v = (v > 20.f) ? v : log1pf(__expf(v));
        out0[(size_t)m * N + n] = v;
      }
    }
  }
}

// ---------------------------------------------------------------------------
__global__ void cvt_bf16(const float* __restrict__ in, unsigned short* __restrict__ out, int n) {
  int i = blockIdx.x * blockDim.x + threadIdx.x;
  if (i < n) out[i] = f2bf(in[i]);
}

// ---------------------------------------------------------------------------
// h = LayerNorm(transpose(x)):  x is (B, D, S)  ->  h (B*S, D)  (+ bf16 copy)
// ---------------------------------------------------------------------------
__global__ void trans_ln(const float* __restrict__ x,
                         float* __restrict__ h, unsigned short* __restrict__ hb) {
  int row = blockIdx.x;
  int b   = row >> 11;            // S_ = 2048
  int s   = row & (S_ - 1);
  int t   = threadIdx.x;
  float v = x[((size_t)b * D_ + t) * S_ + s];

  float a = v, q = v * v;
  for (int o = 16; o; o >>= 1) { a += __shfl_xor(a, o, 32); q += __shfl_xor(q, o, 32); }
  __shared__ float sa[8], sq[8];
  int w = t >> 5, ln = t & 31;
  if (ln == 0) { sa[w] = a; sq[w] = q; }
  __syncthreads();
  a = 0.f; q = 0.f;
#pragma unroll
  for (int i = 0; i < 8; i++) { a += sa[i]; q += sq[i]; }
  float mu  = a * (1.f / D_);
  float var = q * (1.f / D_) - mu * mu;
  float o2  = (v - mu) * rsqrtf(var + EPS_);
  size_t idx = (size_t)row * D_ + t;
  h[idx]  = o2;
  hb[idx] = f2bf(o2);
}

// ---------------------------------------------------------------------------
// h = LayerNorm(yin + h) * w + b   (residual + affine LN, in-place on h)
// ---------------------------------------------------------------------------
__global__ void ln_res(const float* __restrict__ yin,
                       float* __restrict__ h,
                       const float* __restrict__ w, const float* __restrict__ bns,
                       unsigned short* __restrict__ hb) {
  int row = blockIdx.x;
  int t   = threadIdx.x;
  size_t idx = (size_t)row * D_ + t;
  float v = yin[idx] + h[idx];

  float a = v, q = v * v;
  for (int o = 16; o; o >>= 1) { a += __shfl_xor(a, o, 32); q += __shfl_xor(q, o, 32); }
  __shared__ float sa[8], sq[8];
  int wv = t >> 5, ln = t & 31;
  if (ln == 0) { sa[wv] = a; sq[wv] = q; }
  __syncthreads();
  a = 0.f; q = 0.f;
#pragma unroll
  for (int i = 0; i < 8; i++) { a += sa[i]; q += sq[i]; }
  float mu  = a * (1.f / D_);
  float var = q * (1.f / D_) - mu * mu;
  float o2  = (v - mu) * rsqrtf(var + EPS_) * w[t] + bns[t];
  h[idx]  = o2;
  hb[idx] = f2bf(o2);
}

// ---------------------------------------------------------------------------
// Depthwise causal conv (K=4) + bias + SiLU
// ---------------------------------------------------------------------------
__global__ void conv_silu(const float* __restrict__ upre,
                          const float* __restrict__ cw, const float* __restrict__ cb,
                          float* __restrict__ u, unsigned short* __restrict__ ub) {
  size_t i = (size_t)blockIdx.x * blockDim.x + threadIdx.x;   // over B*S*D
  int d = (int)(i & (D_ - 1));
  int s = (int)((i / D_) & (S_ - 1));
  float acc = cb[d];
  const float* w4 = cw + d * K_;
#pragma unroll
  for (int k = 0; k < K_; k++) {
    int ss = s - (K_ - 1) + k;
    if (ss >= 0) acc += upre[i + (size_t)(k - (K_ - 1)) * D_] * w4[k];
  }
  float v = acc / (1.f + __expf(-acc));       // silu
  u[i]  = v;
  ub[i] = f2bf(v);
}

// ---------------------------------------------------------------------------
// Chunked parallel selective scan (linear recurrence h_t = a_t h_{t-1} + b_t).
// Pass A (WRITE_Y=false): per chunk, local scan from 0; outputs final local
//   state and total decay product (per b,chunk,d,n).
// Carry kernel: 8 sequential chunk combines per (b,d,n)  ->  carry_in.
// Pass C (WRITE_Y=true): re-scan each chunk seeded with carry_in; fuses the
//   C-reduction (shfl_xor over 16-lane half-waves), +u*D and *silu(r).
// Staging goes through LDS via async global->LDS copies (ASYNCcnt) when the
// toolchain exposes them.
// ---------------------------------------------------------------------------
template<bool WRITE_Y>
__global__ void scan_pass(const float* __restrict__ delta,
                          const float* __restrict__ u,
                          const float* __restrict__ rg,
                          const float* __restrict__ dbc,   // (B*S,48): [dt|B|C]
                          const float* __restrict__ alog,
                          const float* __restrict__ dpar,
                          const float* __restrict__ carry, // pass C input
                          float* __restrict__ hlast,       // pass A output
                          float* __restrict__ pprod,       // pass A output
                          float* __restrict__ y, unsigned short* __restrict__ yb) {
  __shared__ float sd[TCH][16], su[TCH][16], sr[TCH][16],
                   sB[TCH][16], sC[TCH][16], sy[TCH][16];
  int b      = blockIdx.y;
  int dblock = blockIdx.x * 16;
  int c      = blockIdx.z;
  int t  = threadIdx.x;
  int dl = t >> 4;          // local channel 0..15
  int n  = t & 15;          // state index 0..15
  int d  = dblock + dl;

  float Acoef = -__expf(alog[d * N_ + n]);
  float Dp    = dpar[d];
  size_t cidx = ((size_t)(b * NC_ + c) * D_ + d) * N_ + n;
  float hst   = WRITE_Y ? carry[cidx] : 0.f;
  float Pacc  = 1.f;

  const int tstart = c * (S_ / NC_);
  for (int t0 = tstart; t0 < tstart + S_ / NC_; t0 += TCH) {
    for (int i = t; i < TCH * 16; i += 256) {
      int row = i >> 4, col = i & 15;
      size_t gb = (size_t)(b * S_ + t0 + row);
      size_t gi = gb * D_ + dblock + col;
      lds_cp32(&sd[row][col], &delta[gi]);
      lds_cp32(&su[row][col], &u[gi]);
      lds_cp32(&sB[row][col], &dbc[gb * 48 + DT_ + col]);
      if (WRITE_Y) {
        lds_cp32(&sr[row][col], &rg[gi]);
        lds_cp32(&sC[row][col], &dbc[gb * 48 + DT_ + N_ + col]);
      }
    }
    lds_cp_wait();
    __syncthreads();

    for (int tt = 0; tt < TCH; tt++) {
      float dlt = sd[tt][dl];
      float uu  = su[tt][dl];
      float dA  = __expf(dlt * Acoef);
      hst = fmaf(dA, hst, dlt * uu * sB[tt][n]);
      if constexpr (!WRITE_Y) {
        Pacc *= dA;
      } else {
        float acc = hst * sC[tt][n];
        acc += __shfl_xor(acc, 1, 16);
        acc += __shfl_xor(acc, 2, 16);
        acc += __shfl_xor(acc, 4, 16);
        acc += __shfl_xor(acc, 8, 16);
        if (n == 0) {
          float yv = acc + uu * Dp;
          float rr = sr[tt][dl];
          yv *= rr / (1.f + __expf(-rr));     // * silu(r)
          sy[tt][dl] = yv;
        }
      }
    }
    __syncthreads();

    if constexpr (WRITE_Y) {
      for (int i = t; i < TCH * 16; i += 256) {
        int row = i >> 4, col = i & 15;
        size_t gi = ((size_t)(b * S_ + t0 + row)) * D_ + dblock + col;
        float v = sy[row][col];
        y[gi]  = v;
        yb[gi] = f2bf(v);
      }
      __syncthreads();
    }
  }

  if constexpr (!WRITE_Y) {
    hlast[cidx] = hst;
    pprod[cidx] = Pacc;
  }
}

// Sequential combine of the NC_ chunk summaries per (b,d,n).
__global__ void scan_carry(const float* __restrict__ hlast,
                           const float* __restrict__ pprod,
                           float* __restrict__ carry) {
  int i  = blockIdx.x * blockDim.x + threadIdx.x;   // over B*D*N
  int b  = i / (D_ * N_);
  int dn = i - b * (D_ * N_);
  float car = 0.f;
#pragma unroll
  for (int c = 0; c < NC_; c++) {
    size_t idx = (size_t)(b * NC_ + c) * (D_ * N_) + dn;
    carry[idx] = car;
    car = fmaf(pprod[idx], car, hlast[idx]);
  }
}

// ---------------------------------------------------------------------------
extern "C" void kernel_launch(void* const* d_in, const int* in_sizes, int n_in,
                              void* d_out, int out_size, void* d_ws, size_t ws_size,
                              hipStream_t stream) {
  (void)in_sizes; (void)n_in; (void)out_size; (void)ws_size;
  const float* x        = (const float*)d_in[0];
  const float* in_proj  = (const float*)d_in[1];
  const float* conv_w   = (const float*)d_in[2];
  const float* conv_b   = (const float*)d_in[3];
  const float* x_proj   = (const float*)d_in[4];
  const float* dt_proj  = (const float*)d_in[5];
  const float* dt_bias  = (const float*)d_in[6];
  const float* A_log    = (const float*)d_in[7];
  const float* D_param  = (const float*)d_in[8];
  const float* out_proj = (const float*)d_in[9];
  const float* lnw      = (const float*)d_in[10];
  const float* lnb      = (const float*)d_in[11];
  const float* proj_w   = (const float*)d_in[12];
  const float* proj_b   = (const float*)d_in[13];
  float* out = (float*)d_out;

  // Workspace carve-up (256B aligned)
  char* base = (char*)d_ws;
  size_t off = 0;
  auto alloc = [&](size_t bytes) -> void* {
    void* p = base + off;
    off = (off + bytes + 255) & ~(size_t)255;
    return p;
  };
  const size_t BSD  = (size_t)M_ * D_;
  const size_t CHNK = (size_t)B_ * NC_ * D_ * N_;
  float*          h     = (float*)alloc(BSD * 4);
  unsigned short* hb    = (unsigned short*)alloc(BSD * 2);
  float*          upre  = (float*)alloc(BSD * 4);
  float*          rbuf  = (float*)alloc(BSD * 4);
  float*          u     = (float*)alloc(BSD * 4);
  unsigned short* ub    = (unsigned short*)alloc(BSD * 2);
  float*          dbc   = (float*)alloc((size_t)M_ * 48 * 4);
  unsigned short* dbcb  = (unsigned short*)alloc((size_t)M_ * 48 * 2);
  float*          delta = (float*)alloc(BSD * 4);
  float*          y     = (float*)alloc(BSD * 4);
  unsigned short* yb    = (unsigned short*)alloc(BSD * 2);
  float*          ytmp  = (float*)alloc(BSD * 4);
  float*          hlast = (float*)alloc(CHNK * 4);
  float*          pprod = (float*)alloc(CHNK * 4);
  float*          carry = (float*)alloc(CHNK * 4);
  unsigned short* ipwb  = (unsigned short*)alloc((size_t)NL_ * 2 * D_ * D_ * 2);
  unsigned short* xpwb  = (unsigned short*)alloc((size_t)NL_ * 48 * D_ * 2);
  unsigned short* dtwb  = (unsigned short*)alloc((size_t)NL_ * D_ * DT_ * 2);
  unsigned short* opwb  = (unsigned short*)alloc((size_t)NL_ * D_ * D_ * 2);
  unsigned short* pwb   = (unsigned short*)alloc((size_t)D_ * D_ * 2);

  // Convert weights to bf16
  auto cvt = [&](const float* src, unsigned short* dst, int n) {
    cvt_bf16<<<(n + 255) / 256, 256, 0, stream>>>(src, dst, n);
  };
  cvt(in_proj,  ipwb, NL_ * 2 * D_ * D_);
  cvt(x_proj,   xpwb, NL_ * 48 * D_);
  cvt(dt_proj,  dtwb, NL_ * D_ * DT_);
  cvt(out_proj, opwb, NL_ * D_ * D_);
  cvt(proj_w,   pwb,  D_ * D_);

  // Pre-LN + transpose: x (B,D,S) -> h (B*S, D)
  trans_ln<<<M_, 256, 0, stream>>>(x, h, hb);

  const dim3 blk(128);                 // 4 wave32 per block
  const dim3 gN512(8, M_ / 64);        // N=512
  const dim3 gN256(4, M_ / 64);        // N=256
  const dim3 gN48 (1, M_ / 64);        // N=48 (4th wave idles)
  const dim3 gScan(D_ / 16, B_, NC_);  // 512 blocks

  for (int l = 0; l < NL_; l++) {
    // 1) in_proj: xr = h @ ipw^T, split into u_pre / r
    gemm_wmma<1, 256><<<gN512, blk, 0, stream>>>(hb, D_, ipwb + (size_t)l * 2 * D_ * D_,
                                                 2 * D_, nullptr, upre, rbuf, nullptr);
    // 2) depthwise causal conv + SiLU
    conv_silu<<<(unsigned)(BSD / 256), 256, 0, stream>>>(
        upre, conv_w + (size_t)l * D_ * K_, conv_b + (size_t)l * D_, u, ub);
    // 3) x_proj: dbc = u @ xpw^T  (f32 for B/C, bf16 for dt GEMM)
    gemm_wmma<0, 256><<<gN48, blk, 0, stream>>>(ub, D_, xpwb + (size_t)l * 48 * D_,
                                                48, nullptr, dbc, nullptr, dbcb);
    // 4) dt_proj + softplus: delta = softplus(dt @ dtw^T + dtb)   (K=16, zero-padded)
    gemm_wmma<2, 16><<<gN256, blk, 0, stream>>>(dbcb, 48, dtwb + (size_t)l * D_ * DT_,
                                                D_, dt_bias + (size_t)l * D_,
                                                delta, nullptr, nullptr);
    // 5) chunked parallel selective scan
    scan_pass<false><<<gScan, 256, 0, stream>>>(
        delta, u, rbuf, dbc, A_log + (size_t)l * D_ * N_, D_param + (size_t)l * D_,
        nullptr, hlast, pprod, nullptr, nullptr);
    scan_carry<<<(B_ * D_ * N_) / 256, 256, 0, stream>>>(hlast, pprod, carry);
    scan_pass<true><<<gScan, 256, 0, stream>>>(
        delta, u, rbuf, dbc, A_log + (size_t)l * D_ * N_, D_param + (size_t)l * D_,
        carry, nullptr, nullptr, y, yb);
    // 6) out_proj
    gemm_wmma<0, 256><<<gN256, blk, 0, stream>>>(yb, D_, opwb + (size_t)l * D_ * D_,
                                                 D_, nullptr, ytmp, nullptr, nullptr);
    // 7) residual + LayerNorm (in place on h)
    ln_res<<<M_, 256, 0, stream>>>(ytmp, h, lnw + (size_t)l * D_, lnb + (size_t)l * D_, hb);
  }

  // Final projection: out = h @ proj_w^T + proj_b
  gemm_wmma<0, 256><<<gN256, blk, 0, stream>>>(hb, D_, pwb, D_, proj_b,
                                               out, nullptr, nullptr);
}
